// SamplesAxialAttention_37546604101760
// MI455X (gfx1250) — matrix operand order repaired and checked by reading, not compiled, not run on hardware
//
#include <hip/hip_runtime.h>
#include <hip/hip_bf16.h>

// MI455X / gfx1250 fused axial-samples-attention.
// bf16 WMMA (v_wmma_f32_16x16x32_bf16) for the three GEMMs, fp32 VALU for
// LayerNorm stats / softmax / 4x4 attention combine. Memory-bound design:
// one read of x (134MB), one write of out (134MB) ~= 11.5us @ 23.3 TB/s.

typedef __attribute__((ext_vector_type(16))) __bf16        v16bf;
typedef __attribute__((ext_vector_type(8)))  float         v8f;
typedef __attribute__((ext_vector_type(4)))  unsigned int  v4u;

union FragCast { v16bf bf; v4u q[2]; };
union BF4      { unsigned short h[4]; unsigned long long u64; };

#define DIM     128
#define HEADS   4
#define HID     32
#define QKV_DIM 384
#define SDIM    256
#define POS     16            // spatial positions per block
#define MROWS   64            // POS * 4 samples
#define LDA     136           // bf16 stride of A/O tiles in LDS (conflict-free)
#define LDQK    264           // bf16 stride of QK tile in LDS

// Native RNE float->bf16 (lets the backend use the hardware cvt path
// instead of a 4-op integer emulation after every WMMA accumulator).
__device__ __forceinline__ unsigned short f2bf(float x) {
    __bf16 h = (__bf16)x;
    return __builtin_bit_cast(unsigned short, h);
}
__device__ __forceinline__ float bf2f(unsigned short h) {
    return __uint_as_float(((unsigned int)h) << 16);
}

// A fragment (16x32 bf16, MxK): lane%16 = row; lane/16 selects K-halves
// {0..7,16..23} vs {8..15,24..31}  -> two contiguous 16B chunks per lane.
__device__ __forceinline__ v16bf load_a_frag(const unsigned short* base, int lda,
                                             int mbase, int kk, int lane) {
    const int half = lane >> 4, mr = lane & 15;
    const unsigned short* p = base + (mbase + mr) * lda + kk + 8 * half;
    FragCast f;
    f.q[0] = *reinterpret_cast<const v4u*>(p);
    f.q[1] = *reinterpret_cast<const v4u*>(p + 16);
    return f.bf;
}
// B fragment (32x16 bf16, KxN): lane%16 = column; lane/16 selects K-halves
// {0..7,16..23} vs {8..15,24..31} -> two 16B chunks per lane from a row-major W[N][K] matrix.
__device__ __forceinline__ v16bf load_b_frag(const unsigned short* W, int ldb,
                                             int nb, int kk, int lane) {
    const int khalf = lane >> 4, n = lane & 15;
    const unsigned short* p = W + (nb + n) * ldb + kk + 8 * khalf;
    FragCast f;
    f.q[0] = *reinterpret_cast<const v4u*>(p);
    f.q[1] = *reinterpret_cast<const v4u*>(p + 16);
    return f.bf;
}

__global__ void convert_weights_kernel(const float* __restrict__ Wqkv,
                                       const float* __restrict__ Wo,
                                       const float* __restrict__ Wg,
                                       unsigned short* __restrict__ ws) {
    const int id = blockIdx.x * blockDim.x + threadIdx.x;
    const int NQ = QKV_DIM * DIM;           // 49152
    const int NO = DIM * DIM;               // 16384
    if (id < NQ)                ws[id] = f2bf(Wqkv[id]);
    else if (id < NQ + NO)      ws[id] = f2bf(Wo[id - NQ]);
    else if (id < NQ + 2 * NO)  ws[id] = f2bf(Wg[id - NQ - NO]);
}

__global__ __launch_bounds__(256) void axial_attn_kernel(
    const float* __restrict__ x, const float* __restrict__ mask,
    const float* __restrict__ norm_w, const float* __restrict__ norm_b,
    const unsigned short* __restrict__ wQkv,
    const unsigned short* __restrict__ wOt,
    const unsigned short* __restrict__ wGt,
    float* __restrict__ out)
{
    __shared__ unsigned short sA[MROWS * LDA];     // normalized x, bf16
    __shared__ unsigned short sQK[MROWS * LDQK];   // q|k per head; reused as O
    __shared__ float sAttn[POS * HEADS * 16];      // 4x4 attn per (pos,head)
    __shared__ float sNw[DIM], sNb[DIM];

    const int tid  = threadIdx.x;
    const int wave = tid >> 5;
    const int lane = tid & 31;
    const int i    = (int)blockIdx.x >> 4;
    const int j0   = ((int)blockIdx.x & 15) * POS;

    if (tid < DIM) sNw[tid] = norm_w[tid];
    else           sNb[tid - DIM] = norm_b[tid - DIM];

    // ---- Phase 1: load x tile + LayerNorm -> bf16 A tile in LDS ----
    {
        const int r    = tid >> 1;          // tile row 0..127 is wrong; 0..127/2
        const int part = tid & 1;           // which 64-col half
        const int mm = r & 3, p = r >> 2;   // sample, position-in-tile
        const size_t gbase =
            (((size_t)mm * SDIM + i) * SDIM + (j0 + p)) * DIM + part * 64;
        const float4* px = reinterpret_cast<const float4*>(x + gbase);
        float4 v[16];
        float s = 0.f, sq = 0.f;
#pragma unroll
        for (int t = 0; t < 16; ++t) {
            v[t] = px[t];
            s  += v[t].x + v[t].y + v[t].z + v[t].w;
            sq += v[t].x * v[t].x + v[t].y * v[t].y +
                  v[t].z * v[t].z + v[t].w * v[t].w;
        }
        s  += __shfl_xor(s, 1);
        sq += __shfl_xor(sq, 1);
        const float mean = s * (1.f / DIM);
        const float var  = sq * (1.f / DIM) - mean * mean;
        const float rstd = rsqrtf(var + 1e-5f);
        __syncthreads();                    // sNw/sNb visible
        unsigned short* dst = sA + r * LDA + part * 64;
#pragma unroll
        for (int t = 0; t < 16; ++t) {
            const int c = part * 64 + t * 4;
            BF4 b;
            b.h[0] = f2bf((v[t].x - mean) * rstd * sNw[c + 0] + sNb[c + 0]);
            b.h[1] = f2bf((v[t].y - mean) * rstd * sNw[c + 1] + sNb[c + 1]);
            b.h[2] = f2bf((v[t].z - mean) * rstd * sNw[c + 2] + sNb[c + 2]);
            b.h[3] = f2bf((v[t].w - mean) * rstd * sNw[c + 3] + sNb[c + 3]);
            *reinterpret_cast<unsigned long long*>(dst + t * 4) = b.u64; // b64 store
        }
    }
    __syncthreads();

    // ---- Phase 2: Q/K projection (WMMA), 2 N-tiles per wave ----
    // qkv column e = 96*hh + t*32 + d (t: 0=q,1=k,2=v). q/k = 16 N-tiles.
    for (int t = 2 * wave; t < 2 * wave + 2; ++t) {
        const int hh = t >> 2, s4 = t & 3;
        const int nb = 96 * hh + 16 * s4;             // e-column base in Wqkv
        v8f acc[4] = {};
        for (int kk = 0; kk < DIM; kk += 32) {
            const v16bf bf = load_b_frag(wQkv, DIM, nb, kk, lane);
#pragma unroll
            for (int mt = 0; mt < 4; ++mt) {
                const v16bf af = load_a_frag(sA, LDA, 16 * mt, kk, lane);
                acc[mt] = __builtin_amdgcn_wmma_f32_16x16x32_bf16(
                    false, af, false, bf, (short)0, acc[mt], false, false);
            }
        }
        const int half = lane >> 4, n16 = lane & 15;
        const int qkcol = 64 * hh + 16 * s4 + n16;    // packed q|k per head
#pragma unroll
        for (int mt = 0; mt < 4; ++mt)
#pragma unroll
            for (int g = 0; g < 8; ++g)
                sQK[(16 * mt + g + 8 * half) * LDQK + qkcol] = f2bf(acc[mt][g]);
    }
    __syncthreads();

    // ---- Phase 3: 4x4 scores + bias + softmax (fp32 VALU) ----
    {
        const int pair = tid >> 2;          // (pos,head): 0..63
        const int mr   = tid & 3;           // query sample row
        const int p = pair >> 2, hh = pair & 3;
        const unsigned short* qrow = sQK + (4 * p + mr) * LDQK + 64 * hh;
        const float scale = 0.17677669529663687f;     // 1/sqrt(32)
        float sc[4];
#pragma unroll
        for (int n = 0; n < 4; ++n) {
            const unsigned short* krow = sQK + (4 * p + n) * LDQK + 64 * hh + 32;
            float d = 0.f;
#pragma unroll
            for (int e = 0; e < HID; ++e) d += bf2f(qrow[e]) * bf2f(krow[e]);
            sc[n] = d * scale;
        }
        // bias depends only on the query row -> cancels in softmax; kept anyway
        const float bias = mask[((size_t)mr * SDIM + i) * SDIM + (j0 + p)];
        float mx = -3.4e38f;
#pragma unroll
        for (int n = 0; n < 4; ++n) { sc[n] += bias; mx = fmaxf(mx, sc[n]); }
        float den = 0.f;
#pragma unroll
        for (int n = 0; n < 4; ++n) { sc[n] = __expf(sc[n] - mx); den += sc[n]; }
        const float inv = 1.f / den;
#pragma unroll
        for (int n = 0; n < 4; ++n)
            sAttn[pair * 16 + mr * 4 + n] = sc[n] * inv;
    }
    __syncthreads();

    // ---- Phase 4: V projection (WMMA) + in-register o = attn @ v ----
    unsigned short* sO = sQK;               // reuse q/k buffer, stride LDA
    {
        const int hh = wave >> 1, sub = wave & 1;
        const int nb = 96 * hh + 64 + 16 * sub;       // v columns of Wqkv
        v8f cv[4] = {};
        for (int kk = 0; kk < DIM; kk += 32) {
            const v16bf bf = load_b_frag(wQkv, DIM, nb, kk, lane);
#pragma unroll
            for (int mt = 0; mt < 4; ++mt) {
                const v16bf af = load_a_frag(sA, LDA, 16 * mt, kk, lane);
                cv[mt] = __builtin_amdgcn_wmma_f32_16x16x32_bf16(
                    false, af, false, bf, (short)0, cv[mt], false, false);
            }
        }
        const int half = lane >> 4, n16 = lane & 15;
        const int ocol = 32 * hh + 16 * sub + n16;    // o column e' = 32*hh + d
#pragma unroll
        for (int mt = 0; mt < 4; ++mt) {
            const int p0 = 4 * mt + 2 * half;         // rows g<4 -> p0, g>=4 -> p0+1
            float o[8];
#pragma unroll
            for (int gb = 0; gb < 2; ++gb) {
                const float* at = &sAttn[((p0 + gb) * 4 + hh) * 16];
#pragma unroll
                for (int m2 = 0; m2 < 4; ++m2) {
                    float a = 0.f;
#pragma unroll
                    for (int n = 0; n < 4; ++n)
                        a += at[m2 * 4 + n] * cv[mt][gb * 4 + n];
                    o[gb * 4 + m2] = a;
                }
            }
#pragma unroll
            for (int g = 0; g < 8; ++g)
                sO[(16 * mt + g + 8 * half) * LDA + ocol] = f2bf(o[g]);
        }
    }
    __syncthreads();

    // ---- Phase 5: output GEMMs (Wo, Wg) + gated epilogue ----
    {
        const int nb = 16 * wave;                     // output d-column tile
        v8f co[4] = {}, cg[4] = {};
        for (int kk = 0; kk < DIM; kk += 32) {
            const v16bf bo = load_b_frag(wOt, DIM, nb, kk, lane);
            const v16bf bg = load_b_frag(wGt, DIM, nb, kk, lane);
#pragma unroll
            for (int mt = 0; mt < 4; ++mt) {
                const v16bf af = load_a_frag(sO, LDA, 16 * mt, kk, lane);
                co[mt] = __builtin_amdgcn_wmma_f32_16x16x32_bf16(
                    false, af, false, bo, (short)0, co[mt], false, false);
                cg[mt] = __builtin_amdgcn_wmma_f32_16x16x32_bf16(
                    false, af, false, bg, (short)0, cg[mt], false, false);
            }
        }
        const int half = lane >> 4, n16 = lane & 15;
        const int d = nb + n16;
#pragma unroll
        for (int mt = 0; mt < 4; ++mt)
#pragma unroll
            for (int g = 0; g < 8; ++g) {
                const int row = 16 * mt + g + 8 * half;
                const int p = row >> 2, mm = row & 3;
                const float vo = co[mt][g], vg = cg[mt][g];
                out[(((size_t)mm * SDIM + i) * SDIM + (j0 + p)) * DIM + d] =
                    vo / (1.f + __expf(-vg));
            }
    }
}

extern "C" void kernel_launch(void* const* d_in, const int* in_sizes, int n_in,
                              void* d_out, int out_size, void* d_ws, size_t ws_size,
                              hipStream_t stream) {
    const float* x      = (const float*)d_in[0];
    const float* mask   = (const float*)d_in[1];
    const float* norm_w = (const float*)d_in[2];
    const float* norm_b = (const float*)d_in[3];
    const float* Wqkv   = (const float*)d_in[4];
    const float* Wo     = (const float*)d_in[5];
    const float* Wg     = (const float*)d_in[6];
    float* out = (float*)d_out;

    unsigned short* ws = (unsigned short*)d_ws;
    const unsigned short* wq = ws;
    const unsigned short* wo = ws + QKV_DIM * DIM;
    const unsigned short* wg = ws + QKV_DIM * DIM + DIM * DIM;

    // (49152 + 16384 + 16384) elements -> 320 blocks of 256
    convert_weights_kernel<<<320, 256, 0, stream>>>(Wqkv, Wo, Wg, ws);

    // 256 i-rows x 16 j-tiles of 16 positions
    axial_attn_kernel<<<dim3(SDIM * (SDIM / POS)), dim3(256), 0, stream>>>(
        x, mask, norm_w, norm_b, wq, wo, wg, out);

    (void)in_sizes; (void)n_in; (void)out_size; (void)ws_size;
}
